// LiquidLayer_6339371729206
// MI455X (gfx1250) — matrix operand order repaired and checked by reading, not compiled, not run on hardware
//
#include <hip/hip_runtime.h>
#include <hip/hip_bf16.h>
#include <math.h>

// Problem constants (from reference)
#define Bz   16
#define Sz   2048
#define Dz   512
#define Hz   512
#define NOSC 8
#define STEPS 5
#define DTc  0.1f
#define KCOUP 2.0f
#define AMPc 0.4f
#define TBASEc 0.8f
#define TWO_PIc 6.283185307179586f

typedef __attribute__((ext_vector_type(16))) _Float16 v16h;
typedef __attribute__((ext_vector_type(8)))  _Float16 v8h;
typedef __attribute__((ext_vector_type(8)))  float    v8f;

union F16x16 { v16h v; v8h h[2]; };

// ---------------------------------------------------------------------------
// Kernel 1: transpose + fp32 -> f16 convert:  WT[n*K + k] = (f16) W[k*N + n]
// ---------------------------------------------------------------------------
__global__ void transpose_cvt_kernel(const float* __restrict__ W,
                                     _Float16* __restrict__ WT) {
  int idx = blockIdx.x * blockDim.x + threadIdx.x;   // over 512*512
  if (idx >= Dz * Hz) return;
  int n = idx % Hz;          // consecutive threads -> consecutive n (coalesced read)
  int k = idx / Hz;
  WT[(size_t)n * Dz + k] = (_Float16)W[(size_t)k * Hz + n];
}

// ---------------------------------------------------------------------------
// Kernel 2: fp32 -> f16 elementwise (for x)
// ---------------------------------------------------------------------------
__global__ void cvt_f16_kernel(const float* __restrict__ src,
                               _Float16* __restrict__ dst, int n) {
  int i = blockIdx.x * blockDim.x + threadIdx.x;
  if (i < n) dst[i] = (_Float16)src[i];
}

// ---------------------------------------------------------------------------
// Kernel 3: ext[t] = mean over (B,D) of x[:,t,:]
// ---------------------------------------------------------------------------
__global__ __launch_bounds__(256) void ext_kernel(const float* __restrict__ x,
                                                  float* __restrict__ ext) {
  __shared__ float red[256];
  int t = blockIdx.x;
  float s = 0.f;
  for (int b = 0; b < Bz; ++b) {
    const float* row = x + ((size_t)b * Sz + t) * Dz;
    for (int d = threadIdx.x; d < Dz; d += 256) s += row[d];
  }
  red[threadIdx.x] = s;
  __syncthreads();
  for (int off = 128; off > 0; off >>= 1) {
    if (threadIdx.x < off) red[threadIdx.x] += red[threadIdx.x + off];
    __syncthreads();
  }
  if (threadIdx.x == 0) ext[t] = red[0] / (float)(Bz * Dz);
}

// ---------------------------------------------------------------------------
// Kernel 4: full Kuramoto trajectory on ONE wave; emits 1/temp for every
// (t, inner-step).  Mean-field identity avoids the N^2 sin.
// ---------------------------------------------------------------------------
__global__ void phases_kernel(const float* __restrict__ freqs,
                              const float* __restrict__ phases0,
                              const float* __restrict__ ext,
                              float* __restrict__ temp_inv) {
  int lane = threadIdx.x;                       // 32 lanes; lanes 0..7 live
  float phi = (lane < NOSC) ? phases0[lane] : 0.f;
  float fr  = (lane < NOSC) ? freqs[lane]   : 0.f;
  float s = sinf(phi), c = cosf(phi);
  float S = s, C = c;
  #pragma unroll
  for (int off = 1; off < 8; off <<= 1) { S += __shfl_xor(S, off, 8); C += __shfl_xor(C, off, 8); }

  for (int t = 0; t < Sz; ++t) {
    float e = ext[t] * 0.1f;
    #pragma unroll
    for (int st = 0; st < STEPS; ++st) {
      // sum_j sin(phi_i - phi_j) = sin(phi_i)*C - cos(phi_i)*S
      float coup = (KCOUP / (float)NOSC) * (s * C - c * S);
      float dph  = fr + coup + e;
      phi += dph * DTc;
      phi -= floorf(phi * (1.f / TWO_PIc)) * TWO_PIc;   // floor-mod 2*pi
      s = sinf(phi); c = cosf(phi);
      S = s; C = c;
      #pragma unroll
      for (int off = 1; off < 8; off <<= 1) { S += __shfl_xor(S, off, 8); C += __shfl_xor(C, off, 8); }
      if (lane == 0) {
        // sin(atan2(S/8, C/8)) = S / sqrt(S^2 + C^2)
        float r = rsqrtf(S * S + C * C + 1e-30f);
        float temp = TBASEc + AMPc * (S * r);
        temp_inv[t * STEPS + st] = 1.f / temp;
      }
    }
  }
}

// ---------------------------------------------------------------------------
// Kernel 5: xwib[t][b][n] = (x_t @ Wi)[b][n] + bi[n] + br[n]   (WMMA f16)
// One wave per 16x16 output tile.  M-rows = 16 batches at a fixed t.
// ---------------------------------------------------------------------------
__global__ __launch_bounds__(256) void xwi_kernel(const _Float16* __restrict__ xh,
                                                  const _Float16* __restrict__ WiT,
                                                  const float* __restrict__ bi,
                                                  const float* __restrict__ br,
                                                  float* __restrict__ xwib) {
  int wave  = (blockIdx.x * blockDim.x + threadIdx.x) >> 5;
  int lane  = threadIdx.x & 31;
  int t     = wave >> 5;            // 0..2047
  int ntile = wave & 31;            // 0..31
  int n0    = ntile * 16;
  int half  = lane >> 4, l16 = lane & 15;
  const _Float16* arow = xh  + ((size_t)l16 * Sz + t) * Dz;       // A: M = batch l16
  const _Float16* brow = WiT + (size_t)(n0 + l16) * Dz;           // B: N = l16
  v8f acc = {};
  #pragma unroll 4
  for (int k0 = 0; k0 < Dz; k0 += 32) {
    F16x16 a, b;
    a.h[0] = *(const v8h*)(arow + k0 + 16 * half);
    a.h[1] = *(const v8h*)(arow + k0 + 16 * half + 8);
    b.h[0] = *(const v8h*)(brow + k0 + 16 * half);
    b.h[1] = *(const v8h*)(brow + k0 + 16 * half + 8);
    acc = __builtin_amdgcn_wmma_f32_16x16x32_f16(false, a.v, false, b.v,
                                                 (short)0, acc, false, false);
  }
  int n = n0 + l16;
  float bias = bi[n] + br[n];
  float* ob = xwib + (size_t)t * (Bz * Hz) + n;
  #pragma unroll
  for (int r = 0; r < 8; ++r) ob[(size_t)(r + 8 * half) * Hz] = acc[r] + bias;
}

// ---------------------------------------------------------------------------
// Kernel 6: the serial recurrence.  ONE workgroup (32 waves / one WGP).
// Each lane OWNS its 8 fp32 h-elements in registers (C/D-fragment mapping is
// fixed across steps), so the only LDS state is the f16 WMMA-A copy of h,
// ping-pong double-buffered -> a single s_barrier per inner step.
// Wave w owns N-slice [16w,16w+16) and does both h@Wr and h@Wt tiles
// (32 v_wmma per wave per inner step).  WrT/WtT stream from L2.
// ---------------------------------------------------------------------------
#define HSTR 520   // padded f16 row stride (halves): multiple of 8, != 0 mod 128
#define HBUF (Bz * HSTR)
__global__ __launch_bounds__(1024) void recur_kernel(
    const _Float16* __restrict__ WrT, const _Float16* __restrict__ WtT,
    const float* __restrict__ bt,     const float* __restrict__ xwib,
    const float* __restrict__ temp_inv,
    _Float16* __restrict__ hs, float* __restrict__ h_final_out) {
  __shared__ __align__(16) _Float16 h16[2 * HBUF];   // ping-pong f16 h
  int tid  = threadIdx.x;
  int wave = tid >> 5, lane = tid & 31;
  int half = lane >> 4, l16 = lane & 15;
  int n0   = wave * 16;
  int n    = n0 + l16;
  for (int i = tid; i < 2 * HBUF; i += 1024) h16[i] = (_Float16)0.f;
  float btv = bt[n];
  const _Float16* brrow = WrT + (size_t)n * Hz;
  const _Float16* btrow = WtT + (size_t)n * Hz;
  float hn[8];
  #pragma unroll
  for (int r = 0; r < 8; ++r) hn[r] = 0.f;
  int cur = 0;
  __syncthreads();

  for (int t = 0; t < Sz; ++t) {
    // hoist all per-t uniform/per-lane loads off the inner-step critical path
    const float* xwb = xwib + (size_t)t * (Bz * Hz) + n;
    float xw[8];
    #pragma unroll
    for (int r = 0; r < 8; ++r) xw[r] = xwb[(size_t)(r + 8 * half) * Hz];
    float tv[STEPS];
    #pragma unroll
    for (int st = 0; st < STEPS; ++st) tv[st] = temp_inv[t * STEPS + st];
    if (t + 1 < Sz) __builtin_prefetch(xwib + (size_t)(t + 1) * (Bz * Hz) + n, 0, 1);

    for (int st = 0; st < STEPS; ++st) {
      const _Float16* hcur = &h16[cur * HBUF];
      v8f accr = {}, acct = {};
      #pragma unroll 4
      for (int k0 = 0; k0 < Hz; k0 += 32) {
        F16x16 a, b1, b2;
        a.h[0]  = *(const v8h*)(hcur + l16 * HSTR + k0 + 16 * half);
        a.h[1]  = *(const v8h*)(hcur + l16 * HSTR + k0 + 16 * half + 8);
        b1.h[0] = *(const v8h*)(brrow + k0 + 16 * half);
        b1.h[1] = *(const v8h*)(brrow + k0 + 16 * half + 8);
        b2.h[0] = *(const v8h*)(btrow + k0 + 16 * half);
        b2.h[1] = *(const v8h*)(btrow + k0 + 16 * half + 8);
        accr = __builtin_amdgcn_wmma_f32_16x16x32_f16(false, a.v, false, b1.v,
                                                      (short)0, accr, false, false);
        acct = __builtin_amdgcn_wmma_f32_16x16x32_f16(false, a.v, false, b2.v,
                                                      (short)0, acct, false, false);
      }
      // update h in registers; publish f16 copy into the OTHER buffer
      int nxt = cur ^ 1;
      _Float16* hnxt = &h16[nxt * HBUF];
      float itemp = tv[st];
      #pragma unroll
      for (int r = 0; r < 8; ++r) {
        int m = r + 8 * half;
        float hv  = hn[r];
        float tau = 1.f / (1.f + __expf(-(acct[r] + btv))) + 0.1f;
        float pre = accr[r] + xw[r];
        float dh  = (tanhf(pre) - hv) / tau;
        float v   = (hv + dh * DTc) * itemp;
        hn[r] = v;
        hnxt[m * HSTR + n] = (_Float16)v;
      }
      cur = nxt;
      // single barrier: waves may not enter step st+1 (which overwrites the
      // old 'cur' buffer) until everyone has finished reading it above.
      __syncthreads();
    }
    // emit hs[t][b][n] as f16 (A-operand of the output GEMM)
    _Float16* hrow = hs + (size_t)t * (Bz * Hz) + n;
    #pragma unroll
    for (int r = 0; r < 8; ++r) hrow[(size_t)(r + 8 * half) * Hz] = (_Float16)hn[r];
  }
  // h_final straight from registers (each lane owns its 8 elements)
  float* hf = h_final_out + n;
  #pragma unroll
  for (int r = 0; r < 8; ++r) hf[(size_t)(r + 8 * half) * Hz] = hn[r];
}

// ---------------------------------------------------------------------------
// Kernel 7: yraw[t][b][d] = x[b][t][d] + (hs @ Wo)[t*16+b][d] + bo[d]  (WMMA)
// ---------------------------------------------------------------------------
__global__ __launch_bounds__(256) void outgemm_kernel(const _Float16* __restrict__ hs,
                                                      const _Float16* __restrict__ WoT,
                                                      const float* __restrict__ bo,
                                                      const float* __restrict__ x,
                                                      float* __restrict__ yraw) {
  int wave  = (blockIdx.x * blockDim.x + threadIdx.x) >> 5;
  int lane  = threadIdx.x & 31;
  int t     = wave >> 5;
  int ntile = wave & 31;
  int n0    = ntile * 16;
  int half  = lane >> 4, l16 = lane & 15;
  const _Float16* arow = hs  + ((size_t)t * Bz + l16) * Hz;  // A: M = batch l16 at time t
  const _Float16* brow = WoT + (size_t)(n0 + l16) * Hz;
  v8f acc = {};
  #pragma unroll 4
  for (int k0 = 0; k0 < Hz; k0 += 32) {
    F16x16 a, b;
    a.h[0] = *(const v8h*)(arow + k0 + 16 * half);
    a.h[1] = *(const v8h*)(arow + k0 + 16 * half + 8);
    b.h[0] = *(const v8h*)(brow + k0 + 16 * half);
    b.h[1] = *(const v8h*)(brow + k0 + 16 * half + 8);
    acc = __builtin_amdgcn_wmma_f32_16x16x32_f16(false, a.v, false, b.v,
                                                 (short)0, acc, false, false);
  }
  int d = n0 + l16;
  float bias = bo[d];
  float* ob = yraw + (size_t)t * (Bz * Dz) + d;
  #pragma unroll
  for (int r = 0; r < 8; ++r) {
    int b = r + 8 * half;
    float resid = x[((size_t)b * Sz + t) * Dz + d];
    ob[(size_t)b * Dz] = acc[r] + bias + resid;
  }
}

// ---------------------------------------------------------------------------
// Kernel 8: layernorm over D, one wave per row, writes y[b][t][d] to d_out
// ---------------------------------------------------------------------------
__global__ __launch_bounds__(256) void layernorm_kernel(const float* __restrict__ yraw,
                                                        const float* __restrict__ gamma,
                                                        const float* __restrict__ beta,
                                                        float* __restrict__ out) {
  int row  = blockIdx.x * 8 + (threadIdx.x >> 5);   // 0..32767  (= t*16 + b)
  int lane = threadIdx.x & 31;
  int t = row >> 4, b = row & 15;
  const float* src = yraw + (size_t)row * Dz;
  float vals[16], s = 0.f;
  #pragma unroll
  for (int i = 0; i < 16; ++i) { vals[i] = src[lane + 32 * i]; s += vals[i]; }
  #pragma unroll
  for (int off = 1; off < 32; off <<= 1) s += __shfl_xor(s, off, 32);
  float mu = s * (1.f / Dz);
  float vs = 0.f;
  #pragma unroll
  for (int i = 0; i < 16; ++i) { float d = vals[i] - mu; vs += d * d; }
  #pragma unroll
  for (int off = 1; off < 32; off <<= 1) vs += __shfl_xor(vs, off, 32);
  float rinv = rsqrtf(vs * (1.f / Dz) + 1e-5f);
  float* dst = out + ((size_t)b * Sz + t) * Dz;
  #pragma unroll
  for (int i = 0; i < 16; ++i) {
    int d = lane + 32 * i;
    dst[d] = (vals[i] - mu) * rinv * gamma[d] + beta[d];
  }
}

// ---------------------------------------------------------------------------
// Host-side orchestration
// ---------------------------------------------------------------------------
extern "C" void kernel_launch(void* const* d_in, const int* in_sizes, int n_in,
                              void* d_out, int out_size, void* d_ws, size_t ws_size,
                              hipStream_t stream) {
  const float* x       = (const float*)d_in[0];
  const float* Wi      = (const float*)d_in[1];
  const float* bi      = (const float*)d_in[2];
  const float* Wr      = (const float*)d_in[3];
  const float* br      = (const float*)d_in[4];
  const float* Wt      = (const float*)d_in[5];
  const float* bt      = (const float*)d_in[6];
  const float* Wo      = (const float*)d_in[7];
  const float* bo      = (const float*)d_in[8];
  const float* gamma   = (const float*)d_in[9];
  const float* beta    = (const float*)d_in[10];
  const float* freqs   = (const float*)d_in[11];
  const float* phases0 = (const float*)d_in[12];

  float* y_out   = (float*)d_out;                       // [B,S,D]
  float* h_final = y_out + (size_t)Bz * Sz * Dz;        // [B,H]

  // Workspace layout (~131 MB total; ws assumed sufficient)
  char* p = (char*)d_ws;
  _Float16* xh   = (_Float16*)p;  p += (size_t)Bz * Sz * Dz * 2;   // 32 MB
  _Float16* WiT  = (_Float16*)p;  p += (size_t)Dz * Hz * 2;
  _Float16* WrT  = (_Float16*)p;  p += (size_t)Dz * Hz * 2;
  _Float16* WtT  = (_Float16*)p;  p += (size_t)Dz * Hz * 2;
  _Float16* WoT  = (_Float16*)p;  p += (size_t)Dz * Hz * 2;
  float*    xwib = (float*)p;     p += (size_t)Sz * Bz * Hz * 4;   // 64 MB (reused as yraw)
  _Float16* hsbuf= (_Float16*)p;  p += (size_t)Sz * Bz * Hz * 2;   // 32 MB
  float*    ext  = (float*)p;     p += (size_t)Sz * 4;
  float*    tinv = (float*)p;     p += (size_t)Sz * STEPS * 4;
  float*    yraw = xwib;          // xwib fully consumed before outgemm writes

  const int NX = Bz * Sz * Dz;

  // Phase 0: precision prep (f16 conversions / weight transposes)
  transpose_cvt_kernel<<<(Dz * Hz + 255) / 256, 256, 0, stream>>>(Wi, WiT);
  transpose_cvt_kernel<<<(Dz * Hz + 255) / 256, 256, 0, stream>>>(Wr, WrT);
  transpose_cvt_kernel<<<(Dz * Hz + 255) / 256, 256, 0, stream>>>(Wt, WtT);
  transpose_cvt_kernel<<<(Dz * Hz + 255) / 256, 256, 0, stream>>>(Wo, WoT);
  cvt_f16_kernel<<<(NX + 255) / 256, 256, 0, stream>>>(x, xh, NX);

  // Phase 1: external forcing + full Kuramoto/temperature trajectory
  ext_kernel<<<Sz, 256, 0, stream>>>(x, ext);
  phases_kernel<<<1, 32, 0, stream>>>(freqs, phases0, ext, tinv);

  // Phase 2: hoisted input projection (parallel WMMA GEMM)
  xwi_kernel<<<(Sz * 32) / 8, 256, 0, stream>>>(xh, WiT, bi, br, xwib);

  // Phase 3: serial liquid recurrence (single WGP, register-resident h,
  //          ping-pong f16 LDS copy, one barrier per inner step, WMMA)
  recur_kernel<<<1, 1024, 0, stream>>>(WrT, WtT, bt, xwib, tinv, hsbuf, h_final);

  // Phase 4: output projection + residual (parallel WMMA GEMM), then layernorm
  outgemm_kernel<<<(Sz * 32) / 8, 256, 0, stream>>>(hsbuf, WoT, bo, x, yraw);
  layernorm_kernel<<<(Sz * Bz) / 8, 256, 0, stream>>>(yraw, gamma, beta, y_out);
}
